// GraphResidualBlock_88261577932941
// MI455X (gfx1250) — compile-verified, matrix operand order checked by reading, not run on hardware
//
#include <hip/hip_runtime.h>
#include <hip/hip_bf16.h>
#include <math.h>

#define NN 50000
#define FF 128
#define HH 128
#define EE 800000
#define EPS 1e-5f

typedef __attribute__((ext_vector_type(16))) _Float16 v16h;
typedef __attribute__((ext_vector_type(8)))  _Float16 h8;
typedef __attribute__((ext_vector_type(8)))  float    v8f;
typedef __attribute__((ext_vector_type(4)))  unsigned int u32x4;

__device__ __forceinline__ h8 habs8(h8 a) {
    u32x4 u = __builtin_bit_cast(u32x4, a);
    u = u & 0x7FFF7FFFu;                 // clear f16 sign bits, packed
    return __builtin_bit_cast(h8, u);
}

__device__ __forceinline__ v16h pack16(h8 lo, h8 hi) {
    v16h r;
#pragma unroll
    for (int i = 0; i < 8; ++i) { r[i] = lo[i]; r[i + 8] = hi[i]; }
    return r;
}

__device__ __forceinline__ v8f wmma16x16x32(v16h a, v16h b, v8f c) {
    return __builtin_amdgcn_wmma_f32_16x16x32_f16(false, a, false, b, (short)0, c,
                                                  false, false);
}

// ---------------------------------------------------------------- conversions
__global__ void f32_to_f16_kernel(const float* __restrict__ s,
                                  _Float16* __restrict__ d, int n) {
    int i = blockIdx.x * 256 + threadIdx.x;
    if (i < n) d[i] = (_Float16)s[i];
}

// ------------------------------------------------------- edge-weight MLP (WMMA)
// grid: (EE/256, 3); block 256 (8 waves). Wave w handles 32 edges (two 16-edge
// A tiles sharing every B fragment -> half the LDS traffic per WMMA, and two
// independent accumulator chains to keep the XDL pipe busy).
__global__ __launch_bounds__(256) void edge_mlp_kernel(
    const _Float16* __restrict__ xh,
    const int* __restrict__ ei0, const int* __restrict__ ei1,
    const int* __restrict__ ei2,
    const _Float16* __restrict__ wa1h,
    const float* __restrict__ ba1, const float* __restrict__ wa2,
    const float* __restrict__ ba2,
    float* __restrict__ ew) {
    const int set = blockIdx.y;
    const int* ei = (set == 0) ? ei0 : ((set == 1) ? ei1 : ei2);

    __shared__ __align__(16) _Float16 sB[HH * FF];  // Wa1[set] as f16, 32 KB
    {
        const u32x4* src = (const u32x4*)(wa1h + (size_t)set * HH * FF);
        u32x4* dst = (u32x4*)sB;
        for (int j = threadIdx.x; j < (HH * FF) / 8; j += 256) dst[j] = src[j];
    }
    __syncthreads();

    const int wave = threadIdx.x >> 5;
    const int lane = threadIdx.x & 31;
    const int m = lane & 15;      // edge-in-tile for A; column for B/C
    const int sel = lane >> 4;    // K-half select
    const long e0 = ((long)blockIdx.x * 8 + wave) * 32;
    const int eA = (int)e0 + m;
    const int eB = (int)e0 + 16 + m;
    const int rA = ei[eA], cA = ei[EE + eA];
    const int rB = ei[eB], cB = ei[EE + eB];

    // A fragments: |x[row]-x[col]| in f16, 16x32 layout per K-step, two tiles
    v16h afA[4], afB[4];
#pragma unroll
    for (int ks = 0; ks < 4; ++ks) {
        const int k0 = ks * 32 + sel * 8;
        const int k1 = k0 + 16;
        h8 a0 = *(const h8*)(xh + (size_t)rA * FF + k0);
        h8 b0 = *(const h8*)(xh + (size_t)cA * FF + k0);
        h8 a1 = *(const h8*)(xh + (size_t)rA * FF + k1);
        h8 b1 = *(const h8*)(xh + (size_t)cA * FF + k1);
        afA[ks] = pack16(habs8(a0 - b0), habs8(a1 - b1));
        h8 p0 = *(const h8*)(xh + (size_t)rB * FF + k0);
        h8 q0 = *(const h8*)(xh + (size_t)cB * FF + k0);
        h8 p1 = *(const h8*)(xh + (size_t)rB * FF + k1);
        h8 q1 = *(const h8*)(xh + (size_t)cB * FF + k1);
        afB[ks] = pack16(habs8(p0 - q0), habs8(p1 - q1));
    }

    float accA[8], accB[8];
#pragma unroll
    for (int i = 0; i < 8; ++i) { accA[i] = 0.f; accB[i] = 0.f; }

#pragma unroll
    for (int nt = 0; nt < 8; ++nt) {
        const int h = nt * 16 + m;  // hidden unit covered by this lane's column
        v8f dA = {}, dB = {};
#pragma unroll
        for (int ks = 0; ks < 4; ++ks) {
            const int k0 = ks * 32 + sel * 8;
            const int k1 = k0 + 16;
            h8 blo = *(const h8*)(sB + h * FF + k0);
            h8 bhi = *(const h8*)(sB + h * FF + k1);
            const v16h bf = pack16(blo, bhi);
            dA = wmma16x16x32(afA[ks], bf, dA);
            dB = wmma16x16x32(afB[ks], bf, dB);
        }
        const float bias = ba1[set * HH + h];
        const float w2 = wa2[set * HH + h];
#pragma unroll
        for (int rr = 0; rr < 8; ++rr) {
            float vA = dA[rr] + bias;
            vA = vA > 0.f ? vA : 0.f;      // relu
            accA[rr] += vA * w2;           // dot with Wa2
            float vB = dB[rr] + bias;
            vB = vB > 0.f ? vB : 0.f;
            accB[rr] += vB * w2;
        }
    }

    // reduce over the 16 lanes of each half-wave (rows 0-7 live in lanes 0-15,
    // rows 8-15 in lanes 16-31) — xor masks 1,2,4,8 stay within each half.
#pragma unroll
    for (int rr = 0; rr < 8; ++rr) {
        float vA = accA[rr];
        vA += __shfl_xor(vA, 1, 32);
        vA += __shfl_xor(vA, 2, 32);
        vA += __shfl_xor(vA, 4, 32);
        vA += __shfl_xor(vA, 8, 32);
        accA[rr] = vA;
        float vB = accB[rr];
        vB += __shfl_xor(vB, 1, 32);
        vB += __shfl_xor(vB, 2, 32);
        vB += __shfl_xor(vB, 4, 32);
        vB += __shfl_xor(vB, 8, 32);
        accB[rr] = vB;
    }

    if (m == 0) {
        const float b2 = ba2[set];
        float* dst = ew + (size_t)set * EE + e0 + sel * 8;
#pragma unroll
        for (int rr = 0; rr < 8; ++rr) {
            float zA = accA[rr] + b2;
            dst[rr] = 1.f / (1.f + __expf(-zA));
            float zB = accB[rr] + b2;
            dst[16 + rr] = 1.f / (1.f + __expf(-zB));
        }
    }
}

// ------------------------------------------------------------- WMMA node GEMM
// Y[M,128] = A_h[M,128] * W_h[128(out),128(in)]^T   (f16 in, f32 out)
// Each wave handles two 16-row tiles sharing B fragments.
__global__ __launch_bounds__(256) void gemm_nt_f16(
    const _Float16* __restrict__ Ah, const _Float16* __restrict__ Wh,
    float* __restrict__ Y, int M) {
    __shared__ __align__(16) _Float16 sB[128 * 128];
    {
        const u32x4* src = (const u32x4*)Wh;
        u32x4* dst = (u32x4*)sB;
        for (int j = threadIdx.x; j < (128 * 128) / 8; j += 256) dst[j] = src[j];
    }
    __syncthreads();

    const int wave = threadIdx.x >> 5;
    const int lane = threadIdx.x & 31;
    const int m = lane & 15;
    const int sel = lane >> 4;
    const int ntiles = (M + 15) / 16;
    const int tp = (blockIdx.x * 8 + wave) * 2;
    if (tp >= ntiles) return;
    const bool has2 = (tp + 1) < ntiles;
    const int row0 = tp * 16 + m;
    const int row1 = has2 ? (tp + 1) * 16 + m : row0;

    v16h afA[4], afB[4];
#pragma unroll
    for (int ks = 0; ks < 4; ++ks) {
        const int k0 = ks * 32 + sel * 8;
        const int k1 = k0 + 16;
        afA[ks] = pack16(*(const h8*)(Ah + (size_t)row0 * 128 + k0),
                         *(const h8*)(Ah + (size_t)row0 * 128 + k1));
        afB[ks] = pack16(*(const h8*)(Ah + (size_t)row1 * 128 + k0),
                         *(const h8*)(Ah + (size_t)row1 * 128 + k1));
    }

#pragma unroll
    for (int nt = 0; nt < 8; ++nt) {
        const int col = nt * 16 + m;
        v8f dA = {}, dB = {};
#pragma unroll
        for (int ks = 0; ks < 4; ++ks) {
            const int k0 = ks * 32 + sel * 8;
            const int k1 = k0 + 16;
            h8 blo = *(const h8*)(sB + col * 128 + k0);
            h8 bhi = *(const h8*)(sB + col * 128 + k1);
            const v16h bf = pack16(blo, bhi);
            dA = wmma16x16x32(afA[ks], bf, dA);
            dB = wmma16x16x32(afB[ks], bf, dB);
        }
#pragma unroll
        for (int rr = 0; rr < 8; ++rr) {
            const int orow0 = tp * 16 + rr + 8 * sel;  // C layout: vgpr rr, halves
            Y[(size_t)orow0 * 128 + col] = dA[rr];
        }
        if (has2) {
#pragma unroll
            for (int rr = 0; rr < 8; ++rr) {
                const int orow1 = (tp + 1) * 16 + rr + 8 * sel;
                Y[(size_t)orow1 * 128 + col] = dB[rr];
            }
        }
    }
}

// -------------------------------------------------------------------- degrees
__global__ void degree_kernel(const int* __restrict__ ei0,
                              const int* __restrict__ ei1,
                              const int* __restrict__ ei2,
                              const float* __restrict__ ew,
                              float* __restrict__ deg) {
    long id = (long)blockIdx.x * 256 + threadIdx.x;
    if (id >= 3L * EE) return;
    int set = (int)(id / EE);
    int e = (int)(id % EE);
    const int* ei = (set == 0) ? ei0 : ((set == 1) ? ei1 : ei2);
    atomicAdd(&deg[set * NN + ei[EE + e]], ew[id]);
}

__global__ void dinv_kernel(float* __restrict__ d) {
    int i = blockIdx.x * 256 + threadIdx.x;
    if (i < 3 * NN) {
        float v = d[i];
        d[i] = (v > 0.f) ? rsqrtf(v) : 0.f;
    }
}

// ------------------- fold symmetric normalization into ew (once, in place) ---
__global__ void norm_kernel(const int* __restrict__ ei0,
                            const int* __restrict__ ei1,
                            const int* __restrict__ ei2,
                            const float* __restrict__ dinv,
                            float* __restrict__ ew) {
    long id = (long)blockIdx.x * 256 + threadIdx.x;
    if (id >= 3L * EE) return;
    int set = (int)(id / EE);
    int e = (int)(id % EE);
    const int* ei = (set == 0) ? ei0 : ((set == 1) ? ei1 : ei2);
    ew[id] *= dinv[set * NN + ei[e]] * dinv[set * NN + ei[EE + e]];
}

// --------------------------------------------------------- scatter-aggregate
// grid: (EE/8, 3); one wave per edge; float4 gather + 4 float atomics per lane.
__global__ __launch_bounds__(256) void scatter_kernel(
    const int* __restrict__ ei0, const int* __restrict__ ei1,
    const int* __restrict__ ei2,
    const float* __restrict__ norm,
    const float* __restrict__ src, float* __restrict__ agg) {
    const int set = blockIdx.y;
    const int* ei = (set == 0) ? ei0 : ((set == 1) ? ei1 : ei2);
    const int wave = threadIdx.x >> 5;
    const int lane = threadIdx.x & 31;
    const int e = blockIdx.x * 8 + wave;
    const int r = ei[e];
    const int c = ei[EE + e];
    const float nrm = norm[(size_t)set * EE + e];
    const float4 v = *((const float4*)(src + (size_t)r * FF) + lane);
    float* dst = agg + (size_t)c * FF + lane * 4;
    atomicAdd(dst + 0, nrm * v.x);
    atomicAdd(dst + 1, nrm * v.y);
    atomicAdd(dst + 2, nrm * v.z);
    atomicAdd(dst + 3, nrm * v.w);
}

// --------------------------------------------- act = relu(agg+bias) + BN stats
__global__ __launch_bounds__(256) void act_stats_kernel(
    const float* __restrict__ agg, const float* __restrict__ bias,
    float* __restrict__ act, float* __restrict__ ssum,
    float* __restrict__ ssq) {
    const int t = threadIdx.x;
    const int ch = t & 127;
    const int half = t >> 7;
    const float b = bias[ch];
    float s = 0.f, q = 0.f;
#pragma unroll 4
    for (int i = 0; i < 32; ++i) {
        int r = blockIdx.x * 64 + half + i * 2;
        if (r < NN) {
            size_t idx = (size_t)r * FF + ch;
            float v = agg[idx] + b;
            v = v > 0.f ? v : 0.f;
            act[idx] = v;
            s += v;
            q += v * v;
        }
    }
    __shared__ float red[256];
    red[t] = s;
    __syncthreads();
    if (half == 0) atomicAdd(&ssum[ch], s + red[t + 128]);
    __syncthreads();
    red[t] = q;
    __syncthreads();
    if (half == 0) atomicAdd(&ssq[ch], q + red[t + 128]);
}

// ---------------------------------------- BN + relu -> f16 (input to 2nd GEMM)
__global__ void bn_relu_h16_kernel(const float* __restrict__ act,
                                   const float* __restrict__ ssum,
                                   const float* __restrict__ ssq,
                                   const float* __restrict__ gamma,
                                   const float* __restrict__ beta,
                                   _Float16* __restrict__ outh) {
    size_t idx = (size_t)blockIdx.x * 256 + threadIdx.x;
    if (idx >= (size_t)NN * FF) return;
    int ch = (int)(idx & 127);
    const float inv_n = 1.f / (float)NN;
    float mean = ssum[ch] * inv_n;
    float var = ssq[ch] * inv_n - mean * mean;
    float v = (act[idx] - mean) * rsqrtf(var + EPS) * gamma[ch] + beta[ch];
    v = v > 0.f ? v : 0.f;
    outh[idx] = (_Float16)v;
}

// ---------------------------------------------- final: BN + residual -> d_out
__global__ void bn_residual_kernel(const float* __restrict__ act,
                                   const float* __restrict__ ssum,
                                   const float* __restrict__ ssq,
                                   const float* __restrict__ gamma,
                                   const float* __restrict__ beta,
                                   const float* __restrict__ x,
                                   float* __restrict__ out) {
    size_t idx = (size_t)blockIdx.x * 256 + threadIdx.x;
    if (idx >= (size_t)NN * FF) return;
    int ch = (int)(idx & 127);
    const float inv_n = 1.f / (float)NN;
    float mean = ssum[ch] * inv_n;
    float var = ssq[ch] * inv_n - mean * mean;
    float v = (act[idx] - mean) * rsqrtf(var + EPS) * gamma[ch] + beta[ch];
    out[idx] = v + x[idx];
}

// ============================================================================
extern "C" void kernel_launch(void* const* d_in, const int* in_sizes, int n_in,
                              void* d_out, int out_size, void* d_ws,
                              size_t ws_size, hipStream_t stream) {
    const float* x    = (const float*)d_in[0];
    const int* ei0    = (const int*)d_in[1];
    const int* ei1    = (const int*)d_in[2];
    const int* ei2    = (const int*)d_in[3];
    const float* Wa1  = (const float*)d_in[4];
    const float* ba1  = (const float*)d_in[5];
    const float* Wa2  = (const float*)d_in[6];
    const float* ba2  = (const float*)d_in[7];
    const float* W0   = (const float*)d_in[8];
    const float* b0   = (const float*)d_in[9];
    const float* W1   = (const float*)d_in[10];
    const float* b1   = (const float*)d_in[11];
    const float* g0   = (const float*)d_in[12];
    const float* be0  = (const float*)d_in[13];
    const float* g1   = (const float*)d_in[14];
    const float* be1  = (const float*)d_in[15];
    float* out = (float*)d_out;

    // ---- workspace carve ----
    float* ew   = (float*)d_ws;                  // 3*EE (ew, then norm in place)
    float* dinv = ew + 3 * (size_t)EE;           // 3*NN (deg, then rsqrt in place)
    float* xw   = dinv + 3 * (size_t)NN;         // NN*FF  (GEMM outputs)
    float* agg  = xw + (size_t)NN * FF;          // NN*FF  (scatter target)
    float* act  = agg + (size_t)NN * FF;         // NN*FF  (post-relu activations)
    float* stat = act + (size_t)NN * FF;         // 512: sum0,sq0,sum1,sq1
    _Float16* xh   = (_Float16*)(stat + 512);    // NN*FF halves
    _Float16* hh   = xh + (size_t)NN * FF;       // NN*FF halves
    _Float16* w0h  = hh + (size_t)NN * FF;       // HH*FF
    _Float16* w1h  = w0h + HH * FF;              // FF*HH
    _Float16* wa1h = w1h + FF * HH;              // 3*HH*FF

    hipMemsetAsync(stat, 0, 512 * sizeof(float), stream);
    hipMemsetAsync(dinv, 0, 3 * NN * sizeof(float), stream);

    // ---- f16 conversions ----
    f32_to_f16_kernel<<<(NN * FF + 255) / 256, 256, 0, stream>>>(x, xh, NN * FF);
    f32_to_f16_kernel<<<(3 * HH * FF + 255) / 256, 256, 0, stream>>>(Wa1, wa1h,
                                                                     3 * HH * FF);
    f32_to_f16_kernel<<<(HH * FF + 255) / 256, 256, 0, stream>>>(W0, w0h, HH * FF);
    f32_to_f16_kernel<<<(FF * HH + 255) / 256, 256, 0, stream>>>(W1, w1h, FF * HH);

    // ---- adjacency learning (WMMA) ----
    edge_mlp_kernel<<<dim3(EE / 256, 3), 256, 0, stream>>>(
        xh, ei0, ei1, ei2, wa1h, ba1, Wa2, ba2, ew);

    // ---- degrees -> dinv; fold normalization into ew ----
    degree_kernel<<<(3 * EE + 255) / 256, 256, 0, stream>>>(ei0, ei1, ei2, ew,
                                                            dinv);
    dinv_kernel<<<(3 * NN + 255) / 256, 256, 0, stream>>>(dinv);
    norm_kernel<<<(3 * EE + 255) / 256, 256, 0, stream>>>(ei0, ei1, ei2, dinv,
                                                          ew);

    const int gemm_blocks = ((NN + 15) / 16 + 15) / 16;  // 2 tiles/wave, 8 waves

    // ---- layer 0: xw = x @ W0^T (WMMA) ----
    gemm_nt_f16<<<gemm_blocks, 256, 0, stream>>>(xh, w0h, xw, NN);

    hipMemsetAsync(agg, 0, (size_t)NN * FF * sizeof(float), stream);
    scatter_kernel<<<dim3(EE / 8, 3), 256, 0, stream>>>(ei0, ei1, ei2, ew, xw,
                                                        agg);
    act_stats_kernel<<<(NN + 63) / 64, 256, 0, stream>>>(agg, b0, act, stat,
                                                         stat + 128);
    bn_relu_h16_kernel<<<(NN * FF + 255) / 256, 256, 0, stream>>>(
        act, stat, stat + 128, g0, be0, hh);

    // ---- layer 1: xw = h @ W1^T (WMMA) ----
    gemm_nt_f16<<<gemm_blocks, 256, 0, stream>>>(hh, w1h, xw, NN);

    hipMemsetAsync(agg, 0, (size_t)NN * FF * sizeof(float), stream);
    scatter_kernel<<<dim3(EE / 8, 3), 256, 0, stream>>>(ei0, ei1, ei2, ew, xw,
                                                        agg);
    act_stats_kernel<<<(NN + 63) / 64, 256, 0, stream>>>(agg, b1, act,
                                                         stat + 256, stat + 384);
    bn_residual_kernel<<<(NN * FF + 255) / 256, 256, 0, stream>>>(
        act, stat + 256, stat + 384, g1, be1, x, out);
}
